// ImprovedCausalFlow_70222715290187
// MI455X (gfx1250) — compile-verified, matrix operand order checked by reading, not run on hardware
//
#include <hip/hip_runtime.h>

typedef __bf16 bf16;
typedef bf16  bf16x8 __attribute__((ext_vector_type(8)));
typedef bf16  v16bf  __attribute__((ext_vector_type(16)));
typedef float v8f    __attribute__((ext_vector_type(8)));

#define B_SZ 2048
#define NDIM 64
#define KLAT 128
#define HID  128
#define DK   (NDIM * KLAT)   // 8192
#define EPS  1e-5f

// ---------------------------------------------------------------------------
// Partial GEMM kernel (no epilogue):
//   P[branch][split][M tile 128][128] = A(bf16, MxK) @ (mask ∘ W)(f32->bf16)
// grid = (M/128, nsplit, 2 branches); 256 threads = 8 waves.
// K is processed in 128-wide rounds; B fragments are staged in LDS in the
// exact WMMA B lane layout, double-buffered (ONE barrier per round). Next
// round's W global loads are issued before this round's 32 WMMAs, and the
// B-fragment ds loads are software-pipelined one tile ahead of the WMMAs.
// ---------------------------------------------------------------------------
__global__ __launch_bounds__(256)
void gemm_part(const bf16* __restrict__ AinS, const bf16* __restrict__ AinT,
               int Kdim, int nsplit,
               const float* __restrict__ Ws, const float* __restrict__ Wt,
               const float* __restrict__ cmask, int stepi,
               float* __restrict__ P)
{
    __shared__ __align__(16) bf16x8 blds[2 * 2048];   // 2 x 32KB B-frag buffers

    const int tid  = threadIdx.x;
    const int lane = tid & 31;
    const int wave = tid >> 5;
    const int sp   = blockIdx.y;
    const int br   = blockIdx.z;

    const int Kc    = Kdim / nsplit;
    const int kBase = sp * Kc;

    const bf16*  Abase = br ? AinT : AinS;
    const float* W     = (br ? Wt : Ws) + (long)kBase * HID;

    const int mBase = blockIdx.x * 128;
    // A fragment (16x32 bf16): lane<16 holds K {0..7,16..23} of row lane&15;
    // lane>=16 holds K {8..15,24..31}.
    const int rowA   = mBase + wave * 16 + (lane & 15);
    const int kh     = (lane >> 4) * 8;
    const bf16* arow = Abase + (long)rowA * Kdim + kBase;

    // B staging: thread t owns (tile = t>>5, slotlane = t&31) for all 4 ksubs
    const int wTile = tid >> 5;
    const int wCol  = wTile * 16 + (lane & 15);
    const int wKh   = (lane >> 4) * 8;
    const bool useMask = (cmask != nullptr);

    v8f zero8 = {0.f, 0.f, 0.f, 0.f, 0.f, 0.f, 0.f, 0.f};
    v8f acc[8];
#pragma unroll
    for (int n = 0; n < 8; ++n) acc[n] = zero8;

    // Load one 128-K round of W fragments into registers (mask hoisted: the
    // causal mask is constant per 128-wide K group == one round).
    auto loadW = [&](int k0, bf16x8* wr) {
        float m = useMask ? cmask[((kBase + k0) >> 7) * NDIM + stepi] : 1.0f;
#pragma unroll
        for (int j = 0; j < 4; ++j) {
#pragma unroll
            for (int e = 0; e < 8; ++e) {
                int ka = k0 + j * 32 + wKh + e;
                int kb = ka + 16;
                wr[j * 2 + 0][e] = (bf16)(W[(long)ka * HID + wCol] * m);
                wr[j * 2 + 1][e] = (bf16)(W[(long)kb * HID + wCol] * m);
            }
        }
    };

    bf16x8 wr[8], wn[8];
    loadW(0, wr);
    int p = 0;
    for (int k0 = 0; k0 < Kc; k0 += 128) {
        // publish this round's fragments
#pragma unroll
        for (int j = 0; j < 4; ++j) {
            int s = ((j * 8 + wTile) * 32 + lane) * 2;
            blds[p * 4096 + s]     = wr[j * 2];
            blds[p * 4096 + s + 1] = wr[j * 2 + 1];
        }
        __syncthreads();

        // issue next round's global W loads early (hide behind 32 WMMAs)
        if (k0 + 128 < Kc) loadW(k0 + 128, wn);
        if (k0 + 128 < Kc) __builtin_prefetch(arow + k0 + 128, 0, 3);

#pragma unroll
        for (int j = 0; j < 4; ++j) {
            bf16x8 alo = *(const bf16x8*)(arow + k0 + j * 32 + kh);
            bf16x8 ahi = *(const bf16x8*)(arow + k0 + j * 32 + 16 + kh);
            v16bf af = __builtin_shufflevector(alo, ahi,
                0, 1, 2, 3, 4, 5, 6, 7, 8, 9, 10, 11, 12, 13, 14, 15);

            // B-fragment pipeline: fetch tile n+1 before WMMA on tile n
            const int sbase = p * 4096 + ((j * 8) * 32 + lane) * 2;
            bf16x8 c0 = blds[sbase];
            bf16x8 c1 = blds[sbase + 1];
#pragma unroll
            for (int n = 0; n < 8; ++n) {
                bf16x8 n0, n1;
                if (n < 7) {
                    n0 = blds[sbase + (n + 1) * 64];
                    n1 = blds[sbase + (n + 1) * 64 + 1];
                }
                v16bf bb = __builtin_shufflevector(c0, c1,
                    0, 1, 2, 3, 4, 5, 6, 7, 8, 9, 10, 11, 12, 13, 14, 15);
                acc[n] = __builtin_amdgcn_wmma_f32_16x16x32_bf16(
                    false, af, false, bb, (short)0, acc[n], false, false);
                if (n < 7) { c0 = n0; c1 = n1; }
            }
        }
        p ^= 1;
#pragma unroll
        for (int q = 0; q < 8; ++q) wr[q] = wn[q];
    }

    // store partial C: lane l vgpr r -> row wave*16 + (l<16? r : r+8), col n*16+(l&15)
    float* Pout = P + (((long)br * nsplit + sp) * B_SZ + mBase) * HID;
    const int rb = wave * 16 + ((lane < 16) ? 0 : 8);
    const int cb = lane & 15;
#pragma unroll
    for (int n = 0; n < 8; ++n)
#pragma unroll
        for (int r = 0; r < 8; ++r)
            Pout[(long)(rb + r) * HID + n * 16 + cb] = acc[n][r];
}

// ---------------------------------------------------------------------------
// Epilogue: sum split partials + bias, then
//   mode 0: LayerNorm(g,be) + LeakyReLU(0.2) -> bf16 H
//   mode 1: s-branch tanh(x)*fs, t-branch x*fs -> f32
// grid = (2048 rows, 2 branches), 128 threads (one per column).
// ---------------------------------------------------------------------------
__global__ __launch_bounds__(128)
void mlp_epilogue(const float* __restrict__ P, int nsplit,
                  const float* __restrict__ bS, const float* __restrict__ bT,
                  const float* __restrict__ gS, const float* __restrict__ gT,
                  const float* __restrict__ beS, const float* __restrict__ beT,
                  const float* __restrict__ fsv, int stepi, int mode,
                  bf16* __restrict__ obS, bf16* __restrict__ obT,
                  float* __restrict__ ofS, float* __restrict__ ofT)
{
    const int row = blockIdx.x, br = blockIdx.y, c = threadIdx.x;
    float x = (br ? bT : bS)[c];
    long base = ((long)br * nsplit) * B_SZ * HID + (long)row * HID + c;
    for (int s = 0; s < nsplit; ++s) x += P[base + (long)s * B_SZ * HID];

    if (mode == 0) {
        __shared__ float s1[128], s2[128];
        s1[c] = x; s2[c] = x * x;
        __syncthreads();
        for (int off = 64; off > 0; off >>= 1) {
            if (c < off) { s1[c] += s1[c + off]; s2[c] += s2[c + off]; }
            __syncthreads();
        }
        float mu  = s1[0] * (1.f / 128.f);
        float var = s2[0] * (1.f / 128.f) - mu * mu;
        float v = (x - mu) * rsqrtf(var + EPS) * (br ? gT : gS)[c] + (br ? beT : beS)[c];
        v = (v > 0.f) ? v : 0.2f * v;                    // LeakyReLU(0.2)
        (br ? obT : obS)[(long)row * HID + c] = (bf16)v;
    } else {
        float fs = fsv[stepi];
        float v = br ? x * fs : tanhf(x) * fs;
        (br ? ofT : ofS)[(long)row * HID + c] = v;
    }
}

// out[:, i, :] = z[:, i, :] * exp(clip(s,-5,5)) + t ;  log_det += sum_k s
__global__ __launch_bounds__(128)
void update_step(const float* __restrict__ z, const float* __restrict__ sbuf,
                 const float* __restrict__ tbuf, float* __restrict__ outF,
                 bf16* __restrict__ outB, float* __restrict__ logdet, int stepi)
{
    __shared__ float red[128];
    int b = blockIdx.x, c = threadIdx.x;
    float s  = sbuf[(long)b * KLAT + c];
    float t  = tbuf[(long)b * KLAT + c];
    float sc = __expf(fminf(fmaxf(s, -5.f), 5.f));
    long idx = (long)b * DK + (long)stepi * KLAT + c;
    float v  = z[idx] * sc + t;
    outF[idx] = v;
    outB[idx] = (bf16)v;
    red[c] = s;
    __syncthreads();
    for (int off = 64; off > 0; off >>= 1) {
        if (c < off) red[c] += red[c + off];
        __syncthreads();
    }
    if (c == 0) logdet[b] += red[0];
}

__global__ void init_out(const float* __restrict__ z, float* __restrict__ outF,
                         bf16* __restrict__ outB, long n)
{
    long i = (long)blockIdx.x * blockDim.x + threadIdx.x;
    if (i < n) { float v = z[i]; outF[i] = v; outB[i] = (bf16)v; }
}

__global__ void init_mask(const float* __restrict__ C, float* __restrict__ cm)
{
    int i = blockIdx.x * blockDim.x + threadIdx.x;
    if (i < NDIM * NDIM) {
        int r = i >> 6, c = i & 63;
        float sg = 1.f / (1.f + __expf(-C[i]));   // temperature = 1.0
        cm[i] = (r == c) ? 0.f : sg;
    }
}

__global__ void init_ld(float* __restrict__ ld)
{
    int i = blockIdx.x * blockDim.x + threadIdx.x;
    if (i < B_SZ) ld[i] = 0.f;
}

struct Params { const float *W1,*b1,*g1,*be1,*W2,*b2,*g2,*be2,*W3,*b3; };

extern "C" void kernel_launch(void* const* d_in, const int* in_sizes, int n_in,
                              void* d_out, int out_size, void* d_ws, size_t ws_size,
                              hipStream_t stream)
{
    const long zN = (long)B_SZ * NDIM * KLAT;       // 16,777,216
    const float *z, *Clog, *fsv;
    int base;
    if ((long)in_sizes[0] == zN) {                   // setup_inputs() insertion order
        z    = (const float*)d_in[0];
        Clog = (const float*)d_in[1];
        fsv  = (const float*)d_in[2];
        base = 3;
    } else {                                         // pytree-sorted flattening
        Clog = (const float*)d_in[0];
        fsv  = (const float*)d_in[1];
        base = 2;
        z    = (const float*)d_in[base + 20];
    }

    auto fill = [&](Params& p, int b) {
        const float* q[10];
        for (int j = 0; j < 10; ++j) q[j] = (const float*)d_in[b + j];
        if (in_sizes[b + 1] == NDIM * HID) {         // W1,b1,g1,be1,W2,b2,g2,be2,W3,b3
            p.W1=q[0]; p.b1=q[1]; p.g1=q[2]; p.be1=q[3]; p.W2=q[4];
            p.b2=q[5]; p.g2=q[6]; p.be2=q[7]; p.W3=q[8]; p.b3=q[9];
        } else {                                     // W1,W2,W3,b1,b2,b3,be1,be2,g1,g2
            p.W1=q[0]; p.W2=q[1]; p.W3=q[2]; p.b1=q[3]; p.b2=q[4];
            p.b3=q[5]; p.be1=q[6]; p.be2=q[7]; p.g1=q[8]; p.g2=q[9];
        }
    };
    Params sp, tp;
    fill(sp, base);
    fill(tp, base + 10);

    float* outF   = (float*)d_out;                   // [B, DIM, K] f32
    float* logdet = outF + zN;                       // [B]
    float* cm     = logdet + B_SZ;                   // [DIM, DIM]

    char* w = (char*)d_ws;
    bf16*  outB = (bf16*)w;  w += (size_t)zN * 2;                       // 32 MB
    float* P    = (float*)w; w += (size_t)2 * 4 * B_SZ * HID * 4;       // 8 MB partials
    bf16*  H1s  = (bf16*)w;  w += (size_t)B_SZ * HID * 2;
    bf16*  H1t  = (bf16*)w;  w += (size_t)B_SZ * HID * 2;
    bf16*  H2s  = (bf16*)w;  w += (size_t)B_SZ * HID * 2;
    bf16*  H2t  = (bf16*)w;  w += (size_t)B_SZ * HID * 2;
    float* sbuf = (float*)w; w += (size_t)B_SZ * HID * 4;
    float* tbuf = (float*)w;

    init_out <<<(int)((zN + 255) / 256), 256, 0, stream>>>(z, outF, outB, zN);
    init_mask<<<16, 256, 0, stream>>>(Clog, cm);
    init_ld  <<<8, 256, 0, stream>>>(logdet);

    const dim3 g1(B_SZ / 128, 4, 2);   // split-K=4 on the big GEMM -> 128 WGs
    const dim3 gs(B_SZ / 128, 1, 2);
    const dim3 ge(B_SZ, 2);
    for (int i = 0; i < NDIM; ++i) {
        const long w1o = (long)i * DK * HID;
        const long w2o = (long)i * HID * HID;
        const int  vo  = i * HID;
        // layer 1: [2048x8192] @ [8192x128], mask folded into W rows
        gemm_part<<<g1, 256, 0, stream>>>(outB, outB, DK, 4,
            sp.W1 + w1o, tp.W1 + w1o, cm, i, P);
        mlp_epilogue<<<ge, 128, 0, stream>>>(P, 4,
            sp.b1 + vo, tp.b1 + vo, sp.g1 + vo, tp.g1 + vo, sp.be1 + vo, tp.be1 + vo,
            fsv, i, 0, H1s, H1t, nullptr, nullptr);
        // layer 2: [2048x128] @ [128x128]
        gemm_part<<<gs, 256, 0, stream>>>(H1s, H1t, HID, 1,
            sp.W2 + w2o, tp.W2 + w2o, nullptr, i, P);
        mlp_epilogue<<<ge, 128, 0, stream>>>(P, 1,
            sp.b2 + vo, tp.b2 + vo, sp.g2 + vo, tp.g2 + vo, sp.be2 + vo, tp.be2 + vo,
            fsv, i, 0, H2s, H2t, nullptr, nullptr);
        // layer 3 + final activation: s = tanh(o)*fs, t = o*fs
        gemm_part<<<gs, 256, 0, stream>>>(H2s, H2t, HID, 1,
            sp.W3 + w2o, tp.W3 + w2o, nullptr, i, P);
        mlp_epilogue<<<ge, 128, 0, stream>>>(P, 1,
            sp.b3 + vo, tp.b3 + vo, nullptr, nullptr, nullptr, nullptr,
            fsv, i, 1, nullptr, nullptr, sbuf, tbuf);
        // row update + log_det accumulate
        update_step<<<B_SZ, 128, 0, stream>>>(z, sbuf, tbuf, outF, outB, logdet, i);
    }
}
